// Encoder_16956530884726
// MI455X (gfx1250) — compile-verified, matrix operand order checked by reading, not run on hardware
//
#include <hip/hip_runtime.h>
#include <hip/hip_bf16.h>

typedef __attribute__((ext_vector_type(16))) __bf16 v16bf;
typedef __attribute__((ext_vector_type(8)))  float  v8f;

#define KB 128   // batch
#define KL 256   // seq len
#define KD 128   // input dim
#define KH 256   // hidden
#define KA 128   // action dim

// ---------------------------------------------------------------------------
// WMMA helpers (bf16 A/B, f32 accum, 16x16 tiles, K in chunks of 32)
// A: row-major fp32 source (global or LDS), converted to bf16 in-register.
//    lane layout: m = lane&15, half = lane>>4; elems j0..7 -> k0+half*8+j,
//    j8..15 -> k0+16+half*8+(j-8).
// B: pre-swizzled bf16 so each lane reads 16 contiguous values (32 B).
// D: v8f, element r -> row (r + 8*half), col (lane&15).
// ---------------------------------------------------------------------------
__device__ __forceinline__ v8f wmma_accum_chunks(const float* Abase, int lda,
                                                 const __bf16* Bsw, int NT, int nt,
                                                 int nchunks, v8f acc) {
  int lane = threadIdx.x & 31;
  int m = lane & 15, half = lane >> 4;
  for (int kc = 0; kc < nchunks; ++kc) {
    const float* r0 = Abase + (size_t)m * lda + kc * 32 + half * 8;
    v16bf a;
#pragma unroll
    for (int j = 0; j < 8; ++j) a[j] = (__bf16)r0[j];
#pragma unroll
    for (int j = 0; j < 8; ++j) a[j + 8] = (__bf16)r0[16 + j];
    const __bf16* bp = Bsw + ((((size_t)kc) * NT + nt) * 32 + lane) * 16;
    v16bf b;
#pragma unroll
    for (int j = 0; j < 16; ++j) b[j] = bp[j];
    acc = __builtin_amdgcn_wmma_f32_16x16x32_bf16(false, a, false, b,
                                                  (short)0, acc, false, false);
  }
  return acc;
}

__device__ __forceinline__ void store_tile(float* dst, int ld, v8f acc) {
  int lane = threadIdx.x & 31;
  int n = lane & 15, half = lane >> 4;
#pragma unroll
  for (int rr = 0; rr < 8; ++rr) dst[(size_t)(rr + 8 * half) * ld + n] = acc[rr];
}

// ---------------------------------------------------------------------------
// Swizzle a KxN fp32 weight into bf16 WMMA-B fragment order.
// frag elem j of lane (half=lane/16, nr=lane%16): j<8 -> kr=half*8+j,
// j>=8 -> kr=16+half*8+(j-8); addr = ((kc*NT+nt)*32+lane)*16+j.
// ---------------------------------------------------------------------------
__global__ void swizzle_kernel(const float* __restrict__ src, __bf16* __restrict__ dst,
                               int K, int N) {
  int idx = blockIdx.x * blockDim.x + threadIdx.x;
  if (idx >= K * N) return;
  int k = idx / N, n = idx % N;
  int kc = k >> 5, kr = k & 31, nt = n >> 4, nr = n & 15;
  int q = kr >> 3, rj = kr & 7;
  int half = q & 1;
  int j = (q >> 1) * 8 + rj;
  int lane = half * 16 + nr;
  int NT = N >> 4;
  dst[(((size_t)kc * NT + nt) * 32 + lane) * 16 + j] = (__bf16)src[(size_t)k * N + n];
}

// ---------------------------------------------------------------------------
// Init: dm0 = mask.T, eos = dm0 * (1 - shift_up(dm0)), apc = 0, flags = 0.
// ---------------------------------------------------------------------------
__global__ void init_kernel(const float* __restrict__ mask, float* apc, float* dmc,
                            float* eosb, float* bsum, int* done) {
  int idx = blockIdx.x * blockDim.x + threadIdx.x;
  if (idx < KL * KB) {
    int t = idx / KB, b = idx % KB;
    float m  = mask[(size_t)b * KL + t];
    float mn = (t + 1 < KL) ? mask[(size_t)b * KL + t + 1] : 0.f;
    dmc[idx]  = m;
    eosb[idx] = m * (1.f - mn);
    apc[idx]  = 0.f;
  }
  if (blockIdx.x == 0 && threadIdx.x < 5) {
    if (threadIdx.x < 4) bsum[threadIdx.x] = 0.f;
    else *done = 0;
  }
}

__global__ void done_kernel(const float* bsum, int p, int* done) {
  if (threadIdx.x == 0 && bsum[p] == 0.f) *done = 1;
}

// ---------------------------------------------------------------------------
// Embed: xc[t][b][:] = x[b][t][:] @ W_emb + b_emb   (B*L rows, K=128, N=256)
// block = (b, t-tile of 16); 8 waves, 16 n-tiles.
// ---------------------------------------------------------------------------
__global__ void __launch_bounds__(256) embed_kernel(const float* __restrict__ x,
                                                    const __bf16* __restrict__ WembSw,
                                                    const float* __restrict__ bias,
                                                    float* __restrict__ xc) {
  __shared__ float Gs[16 * KH];
  int blk = blockIdx.x;
  int b  = blk / (KL / 16);
  int t0 = (blk % (KL / 16)) * 16;
  int wave = threadIdx.x >> 5;
  const float* Abase = x + ((size_t)b * KL + t0) * KD;
  for (int nt = wave; nt < KH / 16; nt += 8) {
    v8f acc = {};
    acc = wmma_accum_chunks(Abase, KD, WembSw, KH / 16, nt, KD / 32, acc);
    store_tile(&Gs[nt * 16], KH, acc);
  }
  __syncthreads();
  int r = threadIdx.x >> 4, sub = threadIdx.x & 15;
  size_t orow = ((size_t)(t0 + r) * KB + b) * KH;
  for (int i = sub; i < KH; i += 16) xc[orow + i] = Gs[r * KH + i] + bias[i];
}

// ---------------------------------------------------------------------------
// Per-pass precompute over all L*B rows:
//   s1 = ln(x@W + b) * g0 + be0  (768 cols)   xa = x@W_action_1 + b_action_1 (128)
// block = 16 rows; 8 waves cover 48 + 8 = 56 n-tiles, K = 256 -> 8 chunks.
// ---------------------------------------------------------------------------
__global__ void __launch_bounds__(256) pre_kernel(const float* __restrict__ xc,
                                                  const __bf16* __restrict__ WSw,
                                                  const __bf16* __restrict__ Wa1Sw,
                                                  const float* __restrict__ bvec,
                                                  const float* __restrict__ ba1,
                                                  const float* __restrict__ gam0,
                                                  const float* __restrict__ bet0,
                                                  float* __restrict__ s1,
                                                  float* __restrict__ xa,
                                                  const int* done, int check_done) {
  if (check_done && *done) return;
  __shared__ float Gs[16 * 896];
  __shared__ float red[16][16], red2[16][16];
  __shared__ float mv[16][2];
  int row0 = blockIdx.x * 16;
  int wave = threadIdx.x >> 5;
  const float* Abase = xc + (size_t)row0 * KH;
  for (int tt = wave; tt < 56; tt += 8) {
    v8f acc = {};
    if (tt < 48) {
      acc = wmma_accum_chunks(Abase, KH, WSw, 48, tt, KH / 32, acc);
      store_tile(&Gs[tt * 16], 896, acc);
    } else {
      acc = wmma_accum_chunks(Abase, KH, Wa1Sw, 8, tt - 48, KH / 32, acc);
      store_tile(&Gs[768 + (tt - 48) * 16], 896, acc);
    }
  }
  __syncthreads();
  int r = threadIdx.x >> 4, sub = threadIdx.x & 15;
  float s = 0.f, q = 0.f;
  for (int i = sub; i < 768; i += 16) {
    float v = Gs[r * 896 + i] + bvec[i];
    Gs[r * 896 + i] = v;
    s += v; q += v * v;
  }
  red[r][sub] = s; red2[r][sub] = q;
  __syncthreads();
  if (sub == 0) {
    float S = 0.f, Q = 0.f;
    for (int j = 0; j < 16; ++j) { S += red[r][j]; Q += red2[r][j]; }
    float mean = S / 768.f;
    float var = Q / 768.f - mean * mean; if (var < 0.f) var = 0.f;
    mv[r][0] = mean;
    mv[r][1] = 1.f / (sqrtf(var + 1e-5f) + 1e-5f);
  }
  __syncthreads();
  float mean = mv[r][0], inv = mv[r][1];
  size_t row = (size_t)row0 + r;
  for (int i = sub; i < 768; i += 16)
    s1[row * 768 + i] = gam0[i] * ((Gs[r * 896 + i] - mean) * inv) + bet0[i];
  for (int i = sub; i < 128; i += 16)
    xa[row * 128 + i] = Gs[r * 896 + 768 + i] + ba1[i];
}

// ---------------------------------------------------------------------------
// Scan: each block owns 16 batch rows, h-state resident in LDS, 256 steps.
// Hot weights (U_action_1 + candidate slice of U) are pinned in LDS to keep
// L2 latency off the serial critical path; gate slice streams from L2.
// ---------------------------------------------------------------------------
__global__ void __launch_bounds__(256) scan_kernel(float* __restrict__ xc,
                                                   const float* __restrict__ s1,
                                                   const float* __restrict__ xa,
                                                   float* __restrict__ apc,
                                                   float* __restrict__ dmc,
                                                   const float* __restrict__ eosb,
                                                   const __bf16* __restrict__ USw,
                                                   const __bf16* __restrict__ Ua1Sw,
                                                   const float* __restrict__ gam1,
                                                   const float* __restrict__ bet1,
                                                   const float* __restrict__ W2,
                                                   const float* __restrict__ b2,
                                                   float* bsum, const int* done,
                                                   int pass, int is_final,
                                                   float* __restrict__ out) {
  if (!is_final && *done) return;
  int b0 = blockIdx.x * 16;
  int tid = threadIdx.x, wave = tid >> 5;
  __shared__ float hS[16 * KH];     // recurrent state
  __shared__ float G2[16 * 512];    // gate preacts -> s (z|r)
  __shared__ float Pp[16 * KA];     // policy preact
  __shared__ float RH[16 * KH];     // r * h
  __shared__ float GC[16 * KH];     // candidate preact
  __shared__ float red[16][16], red2[16][16];
  __shared__ float mv[16][2];
  __shared__ float rowS[16][8];     // 0 act, 1 both, 2 honly, 3 xonly, 4 ap, 5 dm, 6 eos
  __shared__ float carry[16][4];    // 0 a_tm1, 1 dmask_tm1, 2 sdm(next), 3 sem(next)
  __shared__ float bsum_loc;
  // LDS-resident swizzled weights (bf16): U[:, 512:768] slice + U_action_1
  __shared__ __align__(16) __bf16 UcL[8 * 16 * 32 * 16];   // 128 KB
  __shared__ __align__(16) __bf16 Ua1L[8 * 8 * 32 * 16];   //  64 KB

  for (int i = tid; i < 16 * KH; i += 256) hS[i] = 0.f;
  if (tid < 16) { carry[tid][0] = 0.f; carry[tid][1] = 0.f; carry[tid][2] = 1.f; carry[tid][3] = 0.f; }
  if (tid == 0) bsum_loc = 0.f;
  {
    // Ua1: layout-identical contiguous copy (16B vectors)
    const uint4* sp = (const uint4*)Ua1Sw;
    uint4* dp = (uint4*)Ua1L;
    for (int i = tid; i < (8 * 8 * 32 * 16) / 8; i += 256) dp[i] = sp[i];
    // U candidate slice: gather nt = 32..47 out of NT = 48 into local NT = 16
    for (int rowi = tid; rowi < 8 * 16 * 32; rowi += 256) {
      int kc = rowi >> 9;          // /(16*32)
      int rem = rowi & 511;
      int ntl = rem >> 5;
      int ln2 = rem & 31;
      const uint4* s4 = (const uint4*)(USw + (((size_t)kc * 48 + 32 + ntl) * 32 + ln2) * 16);
      uint4* d4 = (uint4*)(UcL + (size_t)rowi * 16);
      d4[0] = s4[0];
      d4[1] = s4[1];
    }
  }
  __syncthreads();

  int r = tid >> 4, sub = tid & 15;
  for (int t = 0; t < KL; ++t) {
    // phase 1: gate GEMM (U gates stream from L2) + policy GEMM (LDS weights)
    for (int tt = wave; tt < 40; tt += 8) {
      v8f acc = {};
      if (tt < 32) {
        acc = wmma_accum_chunks(hS, KH, USw, 48, tt, KH / 32, acc);
        store_tile(&G2[tt * 16], 512, acc);
      } else {
        acc = wmma_accum_chunks(hS, KH, Ua1L, 8, tt - 32, KH / 32, acc);
        store_tile(&Pp[(tt - 32) * 16], KA, acc);
      }
    }
    __syncthreads();
    // phase 2: LN(G2) over 512
    {
      float s = 0.f, q = 0.f;
      for (int i = sub; i < 512; i += 16) { float v = G2[r * 512 + i]; s += v; q += v * v; }
      red[r][sub] = s; red2[r][sub] = q;
    }
    __syncthreads();
    if (sub == 0) {
      float S = 0.f, Q = 0.f;
      for (int j = 0; j < 16; ++j) { S += red[r][j]; Q += red2[r][j]; }
      float mean = S / 512.f;
      float var = Q / 512.f - mean * mean; if (var < 0.f) var = 0.f;
      mv[r][0] = mean;
      mv[r][1] = 1.f / (sqrtf(var + 1e-5f) + 1e-5f);
    }
    __syncthreads();
    {
      float mean = mv[r][0], inv = mv[r][1];
      size_t srow = ((size_t)t * KB + b0 + r) * 768;
      for (int i = sub; i < 512; i += 16) {
        float v = gam1[i] * ((G2[r * 512 + i] - mean) * inv) + bet1[i];
        v += s1[srow + i];
        v = 0.2f * v + 0.5f;
        v = v < 0.f ? 0.f : (v > 1.f ? 1.f : v);   // hard_sigmoid
        G2[r * 512 + i] = v;
        if (i >= 256) RH[r * KH + i - 256] = v * hS[r * KH + i - 256];
      }
      float d0 = 0.f, d1 = 0.f;
      size_t arow = ((size_t)t * KB + b0 + r) * KA;
      for (int i = sub; i < KA; i += 16) {
        float pv = Pp[r * KA + i] + xa[arow + i];
        pv = pv > 0.f ? pv : 0.f;                  // relu
        d0 += pv * W2[2 * i];
        d1 += pv * W2[2 * i + 1];
      }
      red[r][sub] = d0; red2[r][sub] = d1;
      if (tid < 16) {
        int b = b0 + tid;
        rowS[tid][4] = apc[(size_t)t * KB + b];
        rowS[tid][5] = dmc[(size_t)t * KB + b];
        rowS[tid][6] = eosb[(size_t)t * KB + b];
      }
    }
    __syncthreads();
    if (sub == 0) {
      float D0 = 0.f, D1 = 0.f;
      for (int j = 0; j < 16; ++j) { D0 += red[r][j]; D1 += red2[r][j]; }
      float p0 = fminf(expf(D0 + b2[0]), 1000.f);
      float p1 = fminf(expf(D1 + b2[1]), 1000.f);
      float act = (p0 <= p1) ? 1.f : 0.f;
      float ap = rowS[r][4], dm = rowS[r][5];
      float sem = carry[r][3], sdm = carry[r][2];
      if (ap > 0.f) act = 1.f;
      if (is_final) act = 1.f;     // llm = ones on final pass
      if (sem > 0.f) act = 0.f;
      float dmtm1 = carry[r][1];
      float both  = (1.f - ap) * dm * act * dmtm1;
      float honly = dmtm1 * act * (ap + (1.f - ap) * (1.f - dm));
      float xonly = dm * (1.f - ap) * (1.f - act + act * (1.f - dmtm1));
      float dmnew = both + honly + xonly;
      rowS[r][0] = act; rowS[r][1] = both; rowS[r][2] = honly; rowS[r][3] = xonly;
      float a_out = (sdm > 0.f) ? act : carry[r][0];
      carry[r][0] = a_out;
      carry[r][1] = dmnew;
      carry[r][2] = dm;            // sdm for step t+1 = dmask[t]
      carry[r][3] = rowS[r][6];    // sem for step t+1 = eos[t]
      int b = b0 + r;
      if (!is_final) {
        if (t > 0) apc[(size_t)(t - 1) * KB + b] = a_out;  // shifted_a
        if (t == KL - 1) apc[(size_t)(KL - 1) * KB + b] = 0.f;
        dmc[(size_t)t * KB + b] = dmnew;
      }
      atomicAdd(&bsum_loc, both);
    }
    __syncthreads();
    // phase 3: candidate GEMM from r*h, weights in LDS
    for (int tt = wave; tt < 16; tt += 8) {
      v8f acc = {};
      acc = wmma_accum_chunks(RH, KH, UcL, 16, tt, KH / 32, acc);
      store_tile(&GC[tt * 16], KH, acc);
    }
    __syncthreads();
    // phase 4: LN(GC) over 256, tanh, gated combine, h update
    {
      float s = 0.f, q = 0.f;
      for (int i = sub; i < KH; i += 16) { float v = GC[r * KH + i]; s += v; q += v * v; }
      red[r][sub] = s; red2[r][sub] = q;
    }
    __syncthreads();
    if (sub == 0) {
      float S = 0.f, Q = 0.f;
      for (int j = 0; j < 16; ++j) { S += red[r][j]; Q += red2[r][j]; }
      float mean = S / 256.f;
      float var = Q / 256.f - mean * mean; if (var < 0.f) var = 0.f;
      mv[r][0] = mean;
      mv[r][1] = 1.f / (sqrtf(var + 1e-5f) + 1e-5f);
    }
    __syncthreads();
    {
      float mean = mv[r][0], inv = mv[r][1];
      float both = rowS[r][1], honly = rowS[r][2], xonly = rowS[r][3], dm = rowS[r][5];
      size_t srow = ((size_t)t * KB + b0 + r) * 768;
      size_t xrow = ((size_t)t * KB + b0 + r) * KH;
      for (int i = sub; i < KH; i += 16) {
        float g = gam1[512 + i] * ((GC[r * KH + i] - mean) * inv) + bet1[512 + i];
        float hcand = tanhf(s1[srow + 512 + i] + g);
        float z = G2[r * 512 + i];
        float hold = hS[r * KH + i];
        float hmix = z * hold + (1.f - z) * hcand;
        float xt = xc[xrow + i];
        float hnew = both * hmix + honly * hold + xonly * xt;
        float hfin = (dm > 0.f) ? hnew : hold;
        hS[r * KH + i] = hfin;
        if (!is_final) xc[xrow + i] = hfin;                      // h_seq in-place
        else if (t == KL - 1) out[(size_t)(b0 + r) * KH + i] = hfin;
      }
    }
    __syncthreads();
  }
  if (tid == 0) atomicAdd(&bsum[pass], bsum_loc);
}

// ---------------------------------------------------------------------------
extern "C" void kernel_launch(void* const* d_in, const int* in_sizes, int n_in,
                              void* d_out, int out_size, void* d_ws, size_t ws_size,
                              hipStream_t stream) {
  const float* x     = (const float*)d_in[0];
  const float* mask  = (const float*)d_in[1];
  // d_in[2] = bucket_size (== KL, fixed)
  const float* Wemb  = (const float*)d_in[3];
  const float* bemb  = (const float*)d_in[4];
  const float* W     = (const float*)d_in[5];
  const float* U     = (const float*)d_in[6];
  const float* bvec  = (const float*)d_in[7];
  const float* Wa1   = (const float*)d_in[8];
  const float* Ua1   = (const float*)d_in[9];
  const float* ba1   = (const float*)d_in[10];
  const float* W2    = (const float*)d_in[11];
  const float* b2    = (const float*)d_in[12];
  const float* gam   = (const float*)d_in[13];
  const float* bet   = (const float*)d_in[14];
  (void)in_sizes; (void)n_in; (void)out_size; (void)ws_size;

  char* base = (char*)d_ws;
  size_t off = 0;
  auto take = [&](size_t bytes) -> char* {
    char* p = base + off;
    off = (off + bytes + 255) & ~(size_t)255;
    return p;
  };
  float*  xc     = (float*)take((size_t)KL * KB * KH * 4);
  float*  s1     = (float*)take((size_t)KL * KB * 768 * 4);
  float*  xa     = (float*)take((size_t)KL * KB * KA * 4);
  float*  apc    = (float*)take((size_t)KL * KB * 4);
  float*  dmc    = (float*)take((size_t)KL * KB * 4);
  float*  eosb   = (float*)take((size_t)KL * KB * 4);
  float*  bsum   = (float*)take(4 * 4);
  int*    done   = (int*)take(4);
  __bf16* WembSw = (__bf16*)take((size_t)KD * KH * 2);
  __bf16* WSw    = (__bf16*)take((size_t)KH * 768 * 2);
  __bf16* USw    = (__bf16*)take((size_t)KH * 768 * 2);
  __bf16* Wa1Sw  = (__bf16*)take((size_t)KH * KA * 2);
  __bf16* Ua1Sw  = (__bf16*)take((size_t)KH * KA * 2);

  swizzle_kernel<<<(KD * KH + 255) / 256, 256, 0, stream>>>(Wemb, WembSw, KD, KH);
  swizzle_kernel<<<(KH * 768 + 255) / 256, 256, 0, stream>>>(W, WSw, KH, 768);
  swizzle_kernel<<<(KH * 768 + 255) / 256, 256, 0, stream>>>(U, USw, KH, 768);
  swizzle_kernel<<<(KH * KA + 255) / 256, 256, 0, stream>>>(Wa1, Wa1Sw, KH, KA);
  swizzle_kernel<<<(KH * KA + 255) / 256, 256, 0, stream>>>(Ua1, Ua1Sw, KH, KA);
  init_kernel<<<(KL * KB + 255) / 256, 256, 0, stream>>>(mask, apc, dmc, eosb, bsum, done);
  embed_kernel<<<KB * (KL / 16), 256, 0, stream>>>(x, WembSw, bemb, xc);

  for (int p = 0; p < 3; ++p) {
    pre_kernel<<<(KL * KB) / 16, 256, 0, stream>>>(xc, WSw, Wa1Sw, bvec, ba1,
                                                   gam, bet, s1, xa, done, 1);
    scan_kernel<<<KB / 16, 256, 0, stream>>>(xc, s1, xa, apc, dmc, eosb, USw, Ua1Sw,
                                             gam + 768, bet + 768, W2, b2,
                                             bsum, done, p, 0, (float*)d_out);
    done_kernel<<<1, 32, 0, stream>>>(bsum, p, done);
  }
  // final horizontal pass with llm = ones, writes h_seq[-1] -> d_out
  pre_kernel<<<(KL * KB) / 16, 256, 0, stream>>>(xc, WSw, Wa1Sw, bvec, ba1,
                                                 gam, bet, s1, xa, done, 0);
  scan_kernel<<<KB / 16, 256, 0, stream>>>(xc, s1, xa, apc, dmc, eosb, USw, Ua1Sw,
                                           gam + 768, bet + 768, W2, b2,
                                           bsum, done, 3, 1, (float*)d_out);
}
